// PostJointNet_9440338117363
// MI455X (gfx1250) — matrix operand order repaired and checked by reading, not compiled, not run on hardware
//
#include <hip/hip_runtime.h>
#include <hip/hip_bf16.h>
#include <stdint.h>

typedef __bf16 bf16_t;
typedef __attribute__((ext_vector_type(16))) __bf16 v16bf;
typedef __attribute__((ext_vector_type(8)))  __bf16 v8bf;
typedef __attribute__((ext_vector_type(8)))  float   v8f;
typedef __attribute__((ext_vector_type(4)))  unsigned int v4u;
typedef __attribute__((ext_vector_type(8)))  int     v8i;
typedef __attribute__((ext_vector_type(4)))  int     v4i;

#define D_IN            256
#define WAVES_PER_BLOCK 8
#define GROUP_M         32      // edges per wave-pair group
#define EDGES_PER_BLOCK 128     // 4 groups
#define PANEL_STRIDE    40      // 32 bf16 + 8 pad -> 20 words/row (conflict-free b128)
#define HBUF_STRIDE     264     // 256 + 8 pad -> 132 words/row (conflict-free b128)

#if defined(__HIP_DEVICE_COMPILE__) && __has_builtin(__builtin_amdgcn_tensor_load_to_lds)
#define HAVE_TDM 1
#warning "CDNA5: tensor_load_to_lds builtin path ENABLED on device pass"
#else
#define HAVE_TDM 0
#endif

// ---------------------------------------------------------------------------
__global__ void cvt_f32_to_bf16(const float* __restrict__ in,
                                bf16_t* __restrict__ out, long long n) {
    long long i = (long long)blockIdx.x * blockDim.x + threadIdx.x;
    long long stride = (long long)gridDim.x * blockDim.x;
    for (; i < n; i += stride) out[i] = (bf16_t)in[i];
}

__device__ inline v16bf frag16(v8bf lo, v8bf hi) {
    v16bf r;
#pragma unroll
    for (int i = 0; i < 8; ++i) { r[i] = lo[i]; r[i + 8] = hi[i]; }
    return r;
}

// B fragment (32x16) for output tile: rows n0..n0+15 of the LDS weight k-panel.
__device__ inline v16bf load_bfrag(const bf16_t* panel, int n0, int le, bool hiL) {
    const bf16_t* wp = panel + (n0 + le) * PANEL_STRIDE + (hiL ? 8 : 0);
    return frag16(*(const v8bf*)wp, *(const v8bf*)(wp + 16));
}

// A fragment (16x32) from a row-major bf16 row pointer at K-offset lo.
__device__ inline v16bf load_afrag(const bf16_t* rp, int lo) {
    return frag16(*(const v8bf*)(rp + lo), *(const v8bf*)(rp + lo + 16));
}

// Stage a 256x32 bf16 weight k-panel (rows padded to PANEL_STRIDE) into LDS.
__device__ inline void stage_panel(const bf16_t* gsrc, int kdim, bf16_t* lpanel) {
#if HAVE_TDM
    if (threadIdx.x < 32) {
        unsigned long long ga = (unsigned long long)(uintptr_t)gsrc;
        unsigned lds_off = (unsigned)(uintptr_t)lpanel;
        v4u g0;
        g0[0] = 1u;                                   // count=1 user descriptor
        g0[1] = lds_off;                              // lds_addr
        g0[2] = (unsigned)(ga & 0xFFFFFFFFu);
        g0[3] = (unsigned)((ga >> 32) & 0x1FFFFFFu) | (2u << 30);  // type=2
        v8i g1;
        g1[0] = (1 << 16)      // data_size = 2 B
              | (1 << 20)      // pad_enable
              | (3 << 22)      // pad_interval: 16 dwords
              | (3 << 25);     // pad_amount:   4 dwords  (64 B -> 80 B rows)
        g1[1] = (int)((unsigned)kdim << 16);          // tensor_dim0[15:0]
        g1[2] = (int)(256u << 16);                    // tensor_dim1[15:0]
        g1[3] = (int)(32u  << 16);                    // tile_dim0 = 32
        g1[4] = 256;                                  // tile_dim1 = 256 rows
        g1[5] = kdim;                                 // tensor_dim0_stride[31:0]
        g1[6] = 0;
        g1[7] = 0;
        v4i z4 = {0, 0, 0, 0};
#if __clang_major__ >= 23
        v8i z8 = {0, 0, 0, 0, 0, 0, 0, 0};
        __builtin_amdgcn_tensor_load_to_lds(g0, g1, z4, z4, z8, 0);
#else
        __builtin_amdgcn_tensor_load_to_lds(g0, g1, z4, z4, 0);
#endif
    }
#else
    const int tid = threadIdx.x;
#pragma unroll
    for (int i = 0; i < 4; ++i) {
        int c = tid + i * 256;
        int n = c >> 2, part = c & 3;
        *(v8bf*)(lpanel + n * PANEL_STRIDE + part * 8) =
            *(const v8bf*)(gsrc + (size_t)n * kdim + part * 8);
    }
#endif
}

__device__ inline void panel_wait_and_sync() {
#if HAVE_TDM
    if (threadIdx.x < 32) {
#if __has_builtin(__builtin_amdgcn_s_wait_tensorcnt)
        __builtin_amdgcn_s_wait_tensorcnt(0);
#else
        asm volatile("s_wait_tensorcnt 0x0" ::: "memory");
#endif
    }
#endif
    __syncthreads();
}

// ---------------------------------------------------------------------------
// Fused: gather -> ELU(h@W1.T+b1) -> ELU(.@W2.T+b2) -> .@W3.T+b3
// Wave = 32 edges x 128 outputs (wave pairs split N); B frag feeds 2 WMMAs.
// ---------------------------------------------------------------------------
__global__ __launch_bounds__(256)
void fused_edge_mlp(const bf16_t* __restrict__ xb,
                    const long long* __restrict__ eidx,
                    const bf16_t* __restrict__ W1b, const float* __restrict__ b1,
                    const bf16_t* __restrict__ W2b, const float* __restrict__ b2,
                    const float*  __restrict__ W3,  const float* __restrict__ b3,
                    float* __restrict__ out, int E)
{
    __shared__ __align__(16) bf16_t wpanel[2][256 * PANEL_STRIDE];          // 40 KB
    __shared__ __align__(16) bf16_t hbuf[4][GROUP_M][HBUF_STRIDE];          // 66 KB

    const int wave = threadIdx.x >> 5;
    const int lane = threadIdx.x & 31;
    const int le   = lane & 15;
    const bool hiL = lane >= 16;
    const int rb   = hiL ? 8 : 0;
    const int eg   = wave >> 1;          // edge group (0..3): 32 edges each
    const int nh   = wave & 1;           // N half (0..1): 128 outputs each

    const int edge0g = blockIdx.x * EDGES_PER_BLOCK + eg * GROUP_M;

    // gather pointers for the two 16-edge sub-tiles
    const bf16_t* srcp[2];
    const bf16_t* dstp[2];
#pragma unroll
    for (int a = 0; a < 2; ++a) {
        int e = edge0g + a * 16 + le; if (e > E - 1) e = E - 1;
        srcp[a] = xb + eidx[e] * D_IN;
        dstp[a] = xb + eidx[(long long)E + e] * D_IN;
    }

    v8f acc[2][8];
    const v8f vzero = {};
#pragma unroll
    for (int a = 0; a < 2; ++a)
#pragma unroll
        for (int nt = 0; nt < 8; ++nt) acc[a][nt] = vzero;

    stage_panel(W1b, 512, &wpanel[0][0]);            // W1 k-panel 0

    const int lofs = hiL ? 8 : 0;
    v16bf a0 = load_afrag(srcp[0], lofs);
    v16bf a1 = load_afrag(srcp[1], lofs);

    // ---------------- Layer 1: K = 512 (16 k-panels) -------------------------
#pragma unroll 1
    for (int ks = 0; ks < 16; ++ks) {
        const int cur = ks & 1;
        panel_wait_and_sync();                        // panel ks visible
        if (ks + 1 < 16) stage_panel(W1b + (ks + 1) * 32, 512, &wpanel[cur ^ 1][0]);
        else             stage_panel(W2b,               256, &wpanel[cur ^ 1][0]);

        // prefetch A fragments for next k-step (L2 gather latency cover)
        v16bf a0n = a0, a1n = a1;
        if (ks + 1 < 16) {
            const int kb2 = (ks + 1) * 32;
            const int lo2 = (kb2 & 255) + lofs;
            a0n = load_afrag((kb2 < D_IN) ? srcp[0] : dstp[0], lo2);
            a1n = load_afrag((kb2 < D_IN) ? srcp[1] : dstp[1], lo2);
        }

        const bf16_t* panel = &wpanel[cur][0];
        const int nbase = nh * 128;
        v16bf br[4];                                  // 4-slot B ring
#pragma unroll
        for (int i = 0; i < 4; ++i) br[i] = load_bfrag(panel, nbase + i * 16, le, hiL);
#pragma unroll
        for (int nt = 0; nt < 8; ++nt) {
            v16bf b = br[nt & 3];
            acc[0][nt] = __builtin_amdgcn_wmma_f32_16x16x32_bf16(
                false, a0, false, b, (short)0, acc[0][nt], false, false);
            acc[1][nt] = __builtin_amdgcn_wmma_f32_16x16x32_bf16(
                false, a1, false, b, (short)0, acc[1][nt], false, false);
            if (nt + 4 < 8)
                br[nt & 3] = load_bfrag(panel, nbase + (nt + 4) * 16, le, hiL);
        }
        __syncthreads();                              // done reading panel ks
        a0 = a0n; a1 = a1n;
    }

    // bias + ELU -> bf16 -> LDS (C layout -> A layout transpose)
#pragma unroll
    for (int nt = 0; nt < 8; ++nt) {
        const int n = nh * 128 + nt * 16 + le;
        const float bias = b1[n];
#pragma unroll
        for (int a = 0; a < 2; ++a)
#pragma unroll
            for (int v = 0; v < 8; ++v) {
                float x = acc[a][nt][v] + bias;
                float y = (x > 0.f) ? x : (__expf(x) - 1.f);
                hbuf[eg][a * 16 + rb + v][n] = (bf16_t)y;
            }
    }

    // ---------------- Layer 2: K = 256 (8 k-panels), A from LDS --------------
#pragma unroll
    for (int a = 0; a < 2; ++a)
#pragma unroll
        for (int nt = 0; nt < 8; ++nt) acc[a][nt] = vzero;
#pragma unroll 1
    for (int ks = 0; ks < 8; ++ks) {
        const int cur = ks & 1;                       // W2 panel ks lives in buf[ks&1]
        panel_wait_and_sync();
        if (ks + 1 < 8) stage_panel(W2b + (ks + 1) * 32, 256, &wpanel[cur ^ 1][0]);

        const int kb = ks * 32;
        const int lo = kb + lofs;
        v16bf a0l = frag16(*(const v8bf*)&hbuf[eg][le][lo],
                           *(const v8bf*)&hbuf[eg][le][lo + 16]);
        v16bf a1l = frag16(*(const v8bf*)&hbuf[eg][16 + le][lo],
                           *(const v8bf*)&hbuf[eg][16 + le][lo + 16]);

        const bf16_t* panel = &wpanel[cur][0];
        const int nbase = nh * 128;
        v16bf br[4];
#pragma unroll
        for (int i = 0; i < 4; ++i) br[i] = load_bfrag(panel, nbase + i * 16, le, hiL);
#pragma unroll
        for (int nt = 0; nt < 8; ++nt) {
            v16bf b = br[nt & 3];
            acc[0][nt] = __builtin_amdgcn_wmma_f32_16x16x32_bf16(
                false, a0l, false, b, (short)0, acc[0][nt], false, false);
            acc[1][nt] = __builtin_amdgcn_wmma_f32_16x16x32_bf16(
                false, a1l, false, b, (short)0, acc[1][nt], false, false);
            if (nt + 4 < 8)
                br[nt & 3] = load_bfrag(panel, nbase + (nt + 4) * 16, le, hiL);
        }
        __syncthreads();
    }
#pragma unroll
    for (int nt = 0; nt < 8; ++nt) {
        const int n = nh * 128 + nt * 16 + le;
        const float bias = b2[n];
#pragma unroll
        for (int a = 0; a < 2; ++a)
#pragma unroll
            for (int v = 0; v < 8; ++v) {
                float x = acc[a][nt][v] + bias;
                float y = (x > 0.f) ? x : (__expf(x) - 1.f);
                hbuf[eg][a * 16 + rb + v][n] = (bf16_t)y;
            }
    }
    __syncthreads();                                  // paired wave's N-half visible

    // ---------------- Layer 3: logits = h2 . W3 + b3 (nh==0 wave, 1 edge/lane)
    if (nh == 0) {
        float s = 0.f;
        const bf16_t* hp = &hbuf[eg][lane][0];
#pragma unroll 1
        for (int k = 0; k < 256; k += 8) {
            v8bf hv = *(const v8bf*)(hp + k);
#pragma unroll
            for (int i = 0; i < 8; ++i) s += (float)hv[i] * W3[k + i];
        }
        const int eo = edge0g + lane;
        if (eo < E) out[eo] = s + b3[0];
    }
}

// ---------------------------------------------------------------------------
extern "C" void kernel_launch(void* const* d_in, const int* in_sizes, int n_in,
                              void* d_out, int out_size, void* d_ws, size_t ws_size,
                              hipStream_t stream) {
    const float* x1 = (const float*)d_in[0];
    const float* x2 = (const float*)d_in[1];
    const long long* eidx = (const long long*)d_in[2];
    const float* W1 = (const float*)d_in[3];
    const float* b1 = (const float*)d_in[4];
    const float* W2 = (const float*)d_in[5];
    const float* b2 = (const float*)d_in[6];
    const float* W3 = (const float*)d_in[7];
    const float* b3 = (const float*)d_in[8];

    const long long n1  = in_sizes[0];
    const long long n2  = in_sizes[1];
    const long long w1n = in_sizes[3];
    const long long w2n = in_sizes[5];
    const int E = in_sizes[2] / 2;

    bf16_t* xb  = (bf16_t*)d_ws;
    bf16_t* W1b = xb + (n1 + n2);
    bf16_t* W2b = W1b + w1n;

    auto blocks_for = [](long long n) {
        long long b = (n + 255) / 256;
        return (int)(b > 4096 ? 4096 : b);
    };
    cvt_f32_to_bf16<<<blocks_for(n1), 256, 0, stream>>>(x1, xb, n1);
    cvt_f32_to_bf16<<<blocks_for(n2), 256, 0, stream>>>(x2, xb + n1, n2);
    cvt_f32_to_bf16<<<blocks_for(w1n), 256, 0, stream>>>(W1, W1b, w1n);
    cvt_f32_to_bf16<<<blocks_for(w2n), 256, 0, stream>>>(W2, W2b, w2n);

    const int nblocks = (E + EDGES_PER_BLOCK - 1) / EDGES_PER_BLOCK;
    fused_edge_mlp<<<nblocks, 256, 0, stream>>>(xb, eidx, W1b, b1, W2b, b2,
                                                W3, b3, (float*)d_out, E);
}
